// RelativeAttention_82102594830987
// MI455X (gfx1250) — matrix operand order repaired and checked by reading, not compiled, not run on hardware
//
#include <hip/hip_runtime.h>
#include <hip/hip_bf16.h>

// ---------------------------------------------------------------------------
// Problem dims (compile-time so pipelined loops fully unroll / rename)
// ---------------------------------------------------------------------------
constexpr int RA_B     = 16384;
constexpr int RA_A     = 4096;
constexpr int RA_D     = 512;
constexpr int RA_H     = 512;
constexpr int RA_CHUNK = 256;

// ---------------------------------------------------------------------------
// Types for CDNA5 WMMA (wave32): V_WMMA_F32_16X16X32_F16
// ---------------------------------------------------------------------------
typedef __attribute__((ext_vector_type(16))) _Float16 v16h;
typedef __attribute__((ext_vector_type(8)))  _Float16 v8h;
typedef __attribute__((ext_vector_type(8)))  float    v8f;

__device__ __forceinline__ v8f vzero8() {
    v8f z;
#pragma unroll
    for (int i = 0; i < 8; ++i) z[i] = 0.0f;
    return z;
}

__device__ __forceinline__ v8f wmma_f16(v16h a, v16h b, v8f c) {
    // (neg_a, A, neg_b, B, c_mod, C, reuse_a, reuse_b)
    return __builtin_amdgcn_wmma_f32_16x16x32_f16(false, a, false, b, (short)0, c,
                                                  false, false);
}

// A fragment (16x32, MxK) from row-major f16 [ld]:
//  lane<16 : row=lane,    K = {0..7, 16..23}
//  lane>=16: row=lane-16, K = {8..15, 24..31}
__device__ __forceinline__ v16h load_a_f16(const _Float16* base, int ld,
                                           int row_base, int k_base, int lane) {
    const int hf = lane >> 4;
    const int r  = lane & 15;
    const _Float16* p = base + (long)(row_base + r) * ld + k_base + hf * 8;
    v8h lo = *(const v8h*)(p);
    v8h hi = *(const v8h*)(p + 16);
    v16h a;
#pragma unroll
    for (int i = 0; i < 8; ++i) { a[i] = lo[i]; a[i + 8] = hi[i]; }
    return a;
}

// Same A fragment layout but source rows are f32 (converted to f16).
__device__ __forceinline__ v16h load_a_f32cvt(const float* base, int ld,
                                              int row_base, int k_base, int lane) {
    const int hf = lane >> 4;
    const int r  = lane & 15;
    const float* p = base + (long)(row_base + r) * ld + k_base + hf * 8;
    v16h a;
#pragma unroll
    for (int i = 0; i < 8; ++i) {
        a[i]     = (_Float16)p[i];
        a[i + 8] = (_Float16)p[16 + i];
    }
    return a;
}

// B fragment (32x16, KxN) from an N-MAJOR f16 matrix (row index = N, contiguous K):
//  lane<16 : col=lane, K = {0..7, 16..23} ; lane>=16: col=lane-16, K = {8..15, 24..31}
// -> two 16-byte loads per lane.
__device__ __forceinline__ v16h load_b_nmajor(const _Float16* base, long ld,
                                              int n_base, int k_base, int lane) {
    const int hf  = lane >> 4;
    const int col = lane & 15;
    const _Float16* p = base + (long)(n_base + col) * ld + k_base + hf * 8;
    v8h lo = *(const v8h*)(p);
    v8h hi = *(const v8h*)(p + 16);
    v16h b;
#pragma unroll
    for (int i = 0; i < 8; ++i) { b[i] = lo[i]; b[i + 8] = hi[i]; }
    return b;
}

// ---------------------------------------------------------------------------
// Kernel 0: transpose + f32->f16 convert.  in: f32 [R][C] -> out: f16 [C][R]
// ---------------------------------------------------------------------------
__global__ __launch_bounds__(256)
void ra_transpose_cvt(const float* __restrict__ in, _Float16* __restrict__ out,
                      int R, int C) {
    long idx = (long)blockIdx.x * blockDim.x + threadIdx.x;
    long n = (long)R * C;
    if (idx >= n) return;
    int r = (int)(idx / C);
    int c = (int)(idx % C);
    out[(long)c * R + r] = (_Float16)in[idx];
}

// ---------------------------------------------------------------------------
// Kernel 1: projection + row L2-norm:  out_f16 = l2norm_rows(X @ W)
//   X : f32 [R][512], Wt : f16 [512][512] N-major, out : f16 [R][512]
// Block = 256 threads (8 waves); 16-row tile per block, wave w owns columns
// [w*64, w*64+64). Fully-unrolled, 1-deep software-pipelined K loop.
// ---------------------------------------------------------------------------
__global__ __launch_bounds__(256)
void ra_project_l2(const float* __restrict__ X, const _Float16* __restrict__ Wt,
                   _Float16* __restrict__ out) {
    __shared__ float ssq[16];
    const int tid  = threadIdx.x;
    const int wave = tid >> 5;
    const int lane = tid & 31;
    const int hf   = lane >> 4;
    const int col  = lane & 15;
    const int m0   = blockIdx.x * 16;
    const int n0   = wave * 64;

    if (tid < 16) ssq[tid] = 0.0f;
    __syncthreads();

    v8f acc[4];
#pragma unroll
    for (int t = 0; t < 4; ++t) acc[t] = vzero8();

    // ---- fully-unrolled software-pipelined K loop (16 steps of K=32) ----
    v16h ax = load_a_f32cvt(X, RA_D, m0, 0, lane);
    v16h bw[4];
#pragma unroll
    for (int t = 0; t < 4; ++t) bw[t] = load_b_nmajor(Wt, RA_D, n0 + t * 16, 0, lane);

#pragma unroll
    for (int ks = 0; ks < RA_D - 32; ks += 32) {
        v16h nax = load_a_f32cvt(X, RA_D, m0, ks + 32, lane);
        v16h nbw[4];
#pragma unroll
        for (int t = 0; t < 4; ++t)
            nbw[t] = load_b_nmajor(Wt, RA_D, n0 + t * 16, ks + 32, lane);
#pragma unroll
        for (int t = 0; t < 4; ++t) acc[t] = wmma_f16(ax, bw[t], acc[t]);
        ax = nax;                       // SSA renames after unroll (no moves)
#pragma unroll
        for (int t = 0; t < 4; ++t) bw[t] = nbw[t];
    }
#pragma unroll
    for (int t = 0; t < 4; ++t) acc[t] = wmma_f16(ax, bw[t], acc[t]);

    // per-lane partial sum-of-squares for the 8 rows this half-wave owns
    float sp[8];
#pragma unroll
    for (int i = 0; i < 8; ++i) sp[i] = 0.0f;
#pragma unroll
    for (int t = 0; t < 4; ++t)
#pragma unroll
        for (int i = 0; i < 8; ++i) sp[i] += acc[t][i] * acc[t][i];
#pragma unroll
    for (int i = 0; i < 8; ++i) atomicAdd(&ssq[hf * 8 + i], sp[i]);
    __syncthreads();

#pragma unroll
    for (int t = 0; t < 4; ++t) {
        const int h = n0 + t * 16 + col;
#pragma unroll
        for (int i = 0; i < 8; ++i) {
            const int row = hf * 8 + i;
            const float sc = 1.0f / fmaxf(sqrtf(ssq[row]), 1e-12f);
            out[(long)(m0 + row) * RA_H + h] = (_Float16)(acc[t][i] * sc);
        }
    }
}

// ---------------------------------------------------------------------------
// Kernel 2: fused attention.
//   qf : f16 [B][512] (L2-normed), kf : f16 [A][512] (L2-normed),
//   vt : f16 [512][A] (values transposed, N-major), out : f32 [B][512]
// Block = 256 threads (8 waves) handles 16 query rows x all 512 outputs.
// q tile staged once into LDS; loop over A in 256-anchor chunks:
//   Phase 1: wave w -> sim for its 32 anchors (unrolled pipelined WMMA over H),
//            bin-round, exp, P -> LDS, per-row denom partials.
//   Phase 2: every wave: acc(16x64) += P(16x256) @ Vt-slice (unrolled pipelined).
// Final: denom reduce (LDS f32 atomics), divide, nontemporal f32 stores.
// ---------------------------------------------------------------------------
__global__ __launch_bounds__(256)
void ra_attention(const _Float16* __restrict__ qf, const _Float16* __restrict__ kf,
                  const _Float16* __restrict__ vt, float* __restrict__ out) {
    __shared__ __attribute__((aligned(32))) _Float16 Pl[16 * RA_CHUNK]; // 8 KB
    __shared__ __attribute__((aligned(32))) _Float16 Qs[16 * RA_H];     // 16 KB
    __shared__ float denomL[16];

    const int tid  = threadIdx.x;
    const int wave = tid >> 5;
    const int lane = tid & 31;
    const int hf   = lane >> 4;
    const int col  = lane & 15;
    const int m0   = blockIdx.x * 16;  // 16 query rows per block
    const int h0   = wave * 64;        // this wave's output-column base

    if (tid < 16) denomL[tid] = 0.0f;

    // ---- stage q tile into LDS (coalesced: each thread copies 64B) ----
    {
        const int r  = tid >> 4;          // 0..15
        const int c0 = (tid & 15) * 32;   // 0..480 step 32
        const _Float16* src = qf + (long)(m0 + r) * RA_H + c0;
        _Float16* dst = &Qs[r * RA_H + c0];
#pragma unroll
        for (int j = 0; j < 4; ++j)
            *(v8h*)(dst + j * 8) = *(const v8h*)(src + j * 8);
    }
    __syncthreads();

    float dpart[8];
#pragma unroll
    for (int i = 0; i < 8; ++i) dpart[i] = 0.0f;

    v8f acc[4];
#pragma unroll
    for (int t = 0; t < 4; ++t) acc[t] = vzero8();

#pragma unroll 1
    for (int a0 = 0; a0 < RA_A; a0 += RA_CHUNK) {
        const int na = a0 + wave * 32;

        // warm caches: next chunk's k rows + this chunk's Vt slice
        if (a0 + RA_CHUNK < RA_A)
            __builtin_prefetch(&kf[(long)(na + RA_CHUNK + col) * RA_H + hf * 256], 0, 3);
        __builtin_prefetch(&vt[(long)(h0 + col) * RA_A + a0 + hf * 128], 0, 3);

        // ---- Phase 1: sim for this wave's 32 anchors (unrolled pipeline) --
        v8f s0 = vzero8(), s1 = vzero8();
        v16h b0 = load_b_nmajor(kf, RA_H, na,      0, lane);
        v16h b1 = load_b_nmajor(kf, RA_H, na + 16, 0, lane);
#pragma unroll
        for (int ks = 0; ks < RA_H - 32; ks += 32) {
            v16h nb0 = load_b_nmajor(kf, RA_H, na,      ks + 32, lane);
            v16h nb1 = load_b_nmajor(kf, RA_H, na + 16, ks + 32, lane);
            v16h aq  = load_a_f16((const _Float16*)Qs, RA_H, 0, ks, lane); // ds
            s0 = wmma_f16(aq, b0, s0);
            s1 = wmma_f16(aq, b1, s1);
            b0 = nb0; b1 = nb1;         // SSA renames after unroll
        }
        {
            v16h aq = load_a_f16((const _Float16*)Qs, RA_H, 0, RA_H - 32, lane);
            s0 = wmma_f16(aq, b0, s0);
            s1 = wmma_f16(aq, b1, s1);
        }

        // bin-round (0.05), exp (sim in [-1,1] -> no max subtraction needed)
#pragma unroll
        for (int i = 0; i < 8; ++i) {
            float v0 = rintf(s0[i] * 20.0f) * 0.05f;   // v_rndne_f32
            float v1 = rintf(s1[i] * 20.0f) * 0.05f;
            float e0 = __expf(v0);
            float e1 = __expf(v1);
            dpart[i] += e0 + e1;
            const int row = hf * 8 + i;
            Pl[row * RA_CHUNK + wave * 32 + col]      = (_Float16)e0;
            Pl[row * RA_CHUNK + wave * 32 + 16 + col] = (_Float16)e1;
        }
        __syncthreads();  // P visible to all waves

        // ---- Phase 2: acc += P(16x256) @ Vt slice (unrolled pipeline) -----
        v16h bv[4];
#pragma unroll
        for (int t = 0; t < 4; ++t)
            bv[t] = load_b_nmajor(vt, RA_A, h0 + t * 16, a0, lane);
#pragma unroll
        for (int ks = 0; ks < RA_CHUNK - 32; ks += 32) {
            v16h nbv[4];
#pragma unroll
            for (int t = 0; t < 4; ++t)
                nbv[t] = load_b_nmajor(vt, RA_A, h0 + t * 16, a0 + ks + 32, lane);
            v16h ap = load_a_f16((const _Float16*)Pl, RA_CHUNK, 0, ks, lane); // ds
#pragma unroll
            for (int t = 0; t < 4; ++t) acc[t] = wmma_f16(ap, bv[t], acc[t]);
#pragma unroll
            for (int t = 0; t < 4; ++t) bv[t] = nbv[t];
        }
        {
            v16h ap = load_a_f16((const _Float16*)Pl, RA_CHUNK, 0,
                                 RA_CHUNK - 32, lane);
#pragma unroll
            for (int t = 0; t < 4; ++t) acc[t] = wmma_f16(ap, bv[t], acc[t]);
        }
        __syncthreads();  // done reading P before next chunk overwrites it
    }

    // ---- denominator reduction across waves/lanes --------------------------
#pragma unroll
    for (int i = 0; i < 8; ++i) atomicAdd(&denomL[hf * 8 + i], dpart[i]);
    __syncthreads();

    // streamed once -> nontemporal stores (keep q/k/Vt resident in L2 instead)
#pragma unroll
    for (int t = 0; t < 4; ++t) {
        const int h = h0 + t * 16 + col;
#pragma unroll
        for (int i = 0; i < 8; ++i) {
            const int row = hf * 8 + i;
            __builtin_nontemporal_store(acc[t][i] / denomL[row],
                                        &out[(long)(m0 + row) * RA_H + h]);
        }
    }
}

// ---------------------------------------------------------------------------
// Launch
// ---------------------------------------------------------------------------
extern "C" void kernel_launch(void* const* d_in, const int* in_sizes, int n_in,
                              void* d_out, int out_size, void* d_ws, size_t ws_size,
                              hipStream_t stream) {
    (void)in_sizes; (void)n_in; (void)out_size; (void)ws_size;
    const float* x       = (const float*)d_in[0];  // [B, D]
    const float* anchors = (const float*)d_in[1];  // [A, D]
    const float* W_q     = (const float*)d_in[2];  // [D, H]
    const float* W_k     = (const float*)d_in[3];  // [D, H]
    const float* values  = (const float*)d_in[4];  // [A, H]
    float* out = (float*)d_out;                    // [B, H]

    // Workspace layout (f16): Wt_q 512KB | Wt_k 512KB | Vt 4MB | Kf 4MB | Qf 16MB
    char* ws = (char*)d_ws;
    _Float16* Wt_q = (_Float16*)(ws);
    _Float16* Wt_k = (_Float16*)(ws + (size_t)512 * 1024);
    _Float16* Vt   = (_Float16*)(ws + (size_t)1024 * 1024);
    _Float16* Kf   = (_Float16*)(ws + (size_t)(1024 + 4096) * 1024);
    _Float16* Qf   = (_Float16*)(ws + (size_t)(1024 + 8192) * 1024);

    // N-major f16 copies so all WMMA B-fragments are contiguous 16B loads.
    ra_transpose_cvt<<<(RA_D * RA_H + 255) / 256, 256, 0, stream>>>(W_q, Wt_q, RA_D, RA_H);
    ra_transpose_cvt<<<(RA_D * RA_H + 255) / 256, 256, 0, stream>>>(W_k, Wt_k, RA_D, RA_H);
    ra_transpose_cvt<<<(RA_A * RA_H + 255) / 256, 256, 0, stream>>>(values, Vt, RA_A, RA_H);

    // k = l2norm(anchors @ W_k),  q = l2norm(x @ W_q)   (f16, L2-resident)
    ra_project_l2<<<RA_A / 16, 256, 0, stream>>>(anchors, Wt_k, Kf);
    ra_project_l2<<<RA_B / 16, 256, 0, stream>>>(x, Wt_q, Qf);

    // Fused sim -> bin-round -> exp -> @values -> /denom
    ra_attention<<<RA_B / 16, 256, 0, stream>>>(Qf, Kf, Vt, out);
}
